// Rnn_52493090291841
// MI455X (gfx1250) — matrix operand order repaired and checked by reading, not compiled, not run on hardware
//
#include <hip/hip_runtime.h>
#include <hip/hip_bf16.h>
#include <cstdint>
#include <cstddef>

// ---------- types ----------
typedef __attribute__((ext_vector_type(16))) __bf16         v16bf;
typedef __attribute__((ext_vector_type(8)))  float          v8f;
typedef __attribute__((ext_vector_type(8)))  unsigned short us8;
typedef __attribute__((ext_vector_type(4)))  unsigned int   u32x4;
typedef __attribute__((ext_vector_type(8)))  unsigned int   u32x8;

static constexpr int Bsz = 4096;
static constexpr int Isz = 1024;
static constexpr int Hsz = 2048;

union Frag {
    v16bf v;
    u32x4 q[2];
    us8   h[2];
};

__device__ __forceinline__ unsigned short f2bf(float f) {
    unsigned int u = __float_as_uint(f);
    u = u + 0x7FFFu + ((u >> 16) & 1u);      // round-to-nearest-even
    return (unsigned short)(u >> 16);
}

__device__ __forceinline__ float fast_tanh(float x) {
#if __has_builtin(__builtin_amdgcn_tanhf)
    return __builtin_amdgcn_tanhf(x);        // v_tanh_f32 (gfx1250 trans op)
#elif __has_builtin(__builtin_amdgcn_tanh_f32)
    return __builtin_amdgcn_tanh_f32(x);
#else
    return tanhf(x);
#endif
}

__device__ __forceinline__ void wait_tensorcnt0() {
#if __has_builtin(__builtin_amdgcn_s_wait_tensorcnt)
    __builtin_amdgcn_s_wait_tensorcnt(0);
#else
    asm volatile("s_wait_tensorcnt 0x0" ::: "memory");
#endif
}

// Issue a TDM 2-D tile load: [rows x 32] bf16 tile from a row-major matrix
// (row stride = kStride elements) into LDS with a padded row stride of
// 40 ushorts (pad_interval=3 -> every 16 dwords, pad_amount=3 -> 4 dwords).
// Descriptor layout per CDNA5 ISA.
__device__ __forceinline__ void tdm_load_tileA(const unsigned short* gptr,
                                               unsigned int ldsByteAddr,
                                               int kStride, int rows) {
    unsigned long long ga = (unsigned long long)(const void*)gptr;
    u32x4 g0;
    g0[0] = 1u;                                         // count=1, user mode
    g0[1] = ldsByteAddr;                                // lds_addr
    g0[2] = (unsigned int)ga;                           // global_addr[31:0]
    g0[3] = (unsigned int)((ga >> 32) & 0x1FFFFFFull)   // global_addr[56:32]
          | (2u << 30);                                 // type=2 ("image")
    u32x8 g1;
    g1[0] = (1u << 16)                                  // data_size = 2B
          | (1u << 20)                                  // pad_enable
          | (3u << 22)                                  // pad_interval: 16 dwords
          | (3u << 25);                                 // pad_amount: 4 dwords
    g1[1] = ((unsigned int)kStride & 0xFFFFu) << 16;    // tensor_dim0[15:0]
    g1[2] = ((unsigned int)Bsz & 0xFFFFu) << 16;        // tensor_dim1[15:0]
    g1[3] = 32u << 16;                                  // tile_dim0 = 32 (k)
    g1[4] = (unsigned int)rows;                         // tile_dim1 = rows
    g1[5] = (unsigned int)kStride;                      // tensor_dim0_stride lo
    g1[6] = 0u;
    g1[7] = 0u;
    asm volatile("tensor_load_to_lds %0, %1" :: "s"(g0), "s"(g1) : "memory");
}

// ---------- stage 1: fp32 -> bf16 convert ----------
__global__ void cvt_bf16(const float* __restrict__ src,
                         unsigned short* __restrict__ dst, int n) {
    int i = blockIdx.x * blockDim.x + threadIdx.x;
    int stride = gridDim.x * blockDim.x;
    for (; i < n; i += stride) dst[i] = f2bf(src[i]);
}

// ---------- stage 2: fused dual-GEMM gate cell ----------
// Macro tile: 64 (M) x 128 (N). 256 threads = 8 waves, each wave owns 32x32.
// A tile staged by the Tensor Data Mover; weight tiles staged + transposed by
// the waves with v_perm_b32 packing.
__global__ __launch_bounds__(256)
void cell_kernel(const unsigned short* __restrict__ xbf,
                 const unsigned short* __restrict__ hbf,
                 const unsigned short* __restrict__ wgx,
                 const unsigned short* __restrict__ wgh,
                 const unsigned short* __restrict__ wxh,
                 const unsigned short* __restrict__ whh,
                 const float* __restrict__ b_h,
                 const float* __restrict__ b_g,
                 const float* __restrict__ g_old,
                 const float* __restrict__ h_old,
                 float* __restrict__ h_new,
                 float* __restrict__ g_new,
                 unsigned short* __restrict__ hnew_bf)
{
    // padded row strides (in ushorts) keep 16B alignment and dodge bank conflicts
    __shared__ unsigned short ldsA [64  * 40];  // A tile   [row][k], 64x32 (TDM-filled)
    __shared__ unsigned short ldsBG[128 * 40];  // Bg tile  [n][k] (transposed), 32x128
    __shared__ unsigned short ldsBH[128 * 40];  // Bh tile  [n][k] (transposed), 32x128

    const int t    = threadIdx.x;
    const int lane = t & 31;
    const int wave = t >> 5;
    const int wm   = wave & 1;      // 2 waves along M
    const int wn   = wave >> 1;     // 4 waves along N
    const int lN   = lane & 15;
    const int hf   = lane >> 4;
    const int mBase = blockIdx.x * 64;
    const int nBase = blockIdx.y * 128;

    const unsigned int ldsA_addr = (unsigned int)(size_t)(&ldsA[0]);

    v8f accG[2][2] = {};
    v8f accH[2][2] = {};

    // cooperative weight-load coordinates: each thread owns a k-pair x 8 n
    const int bK = (t >> 4) * 2, bN = (t & 15) * 8;

    for (int phase = 0; phase < 2; ++phase) {
        const unsigned short* aPtr = phase ? hbf : xbf;
        const unsigned short* wG   = phase ? wgh : wgx;
        const unsigned short* wHp  = phase ? whh : wxh;
        const int kLen = phase ? Hsz : Isz;   // activation row stride == kLen

        for (int kt = 0; kt < kLen; kt += 32) {
            // ---- TDM: DMA the 64x32 A tile into LDS (wave 0 issues) ----
            if (wave == 0) {
                tdm_load_tileA(aPtr + (size_t)mBase * kLen + kt,
                               ldsA_addr, kLen, 64);
            }
            // ---- stage both weight tiles, transposing [k][n] -> [n][k] ----
            {
                const unsigned short* s0 = wG + (size_t)(kt + bK) * Hsz + nBase + bN;
                u32x4 lo = *(const u32x4*)s0;
                u32x4 hi = *(const u32x4*)(s0 + Hsz);
                const unsigned short* s1 = wHp + (size_t)(kt + bK) * Hsz + nBase + bN;
                u32x4 lo2 = *(const u32x4*)s1;
                u32x4 hi2 = *(const u32x4*)(s1 + Hsz);
#pragma unroll
                for (int w = 0; w < 4; ++w) {
                    unsigned int e = __builtin_amdgcn_perm(hi[w],  lo[w],  0x05040100u);
                    unsigned int o = __builtin_amdgcn_perm(hi[w],  lo[w],  0x07060302u);
                    *(unsigned int*)&ldsBG[(bN + 2 * w)     * 40 + bK] = e;
                    *(unsigned int*)&ldsBG[(bN + 2 * w + 1) * 40 + bK] = o;
                    unsigned int e2 = __builtin_amdgcn_perm(hi2[w], lo2[w], 0x05040100u);
                    unsigned int o2 = __builtin_amdgcn_perm(hi2[w], lo2[w], 0x07060302u);
                    *(unsigned int*)&ldsBH[(bN + 2 * w)     * 40 + bK] = e2;
                    *(unsigned int*)&ldsBH[(bN + 2 * w + 1) * 40 + bK] = o2;
                }
            }
            if (wave == 0) wait_tensorcnt0();   // A tile resident in LDS
            __syncthreads();

            // ---- fragments + 8 WMMAs ----
            Frag a[2], bg[2], bh[2];
#pragma unroll
            for (int i = 0; i < 2; ++i) {
                int r = wm * 32 + i * 16 + lN;
                a[i].q[0] = *(const u32x4*)&ldsA[r * 40 + hf * 8];
                a[i].q[1] = *(const u32x4*)&ldsA[r * 40 + 16 + hf * 8];
            }
#pragma unroll
            for (int j = 0; j < 2; ++j) {
                int n = wn * 32 + j * 16 + lN;
                bg[j].q[0] = *(const u32x4*)&ldsBG[n * 40 + hf * 8];
                bg[j].q[1] = *(const u32x4*)&ldsBG[n * 40 + 16 + hf * 8];
                bh[j].q[0] = *(const u32x4*)&ldsBH[n * 40 + hf * 8];
                bh[j].q[1] = *(const u32x4*)&ldsBH[n * 40 + 16 + hf * 8];
            }
#pragma unroll
            for (int i = 0; i < 2; ++i)
#pragma unroll
                for (int j = 0; j < 2; ++j) {
                    accG[i][j] = __builtin_amdgcn_wmma_f32_16x16x32_bf16(
                        false, a[i].v, false, bg[j].v, (short)0, accG[i][j], false, false);
                    accH[i][j] = __builtin_amdgcn_wmma_f32_16x16x32_bf16(
                        false, a[i].v, false, bh[j].v, (short)0, accH[i][j], false, false);
                }
            __syncthreads();
        }
    }

    // ---- fused epilogue: tanh + old-gate interpolation ----
#pragma unroll
    for (int i = 0; i < 2; ++i) {
#pragma unroll
        for (int j = 0; j < 2; ++j) {
            int col = nBase + wn * 32 + j * 16 + lN;
            float bgv = b_g[col];
            float bhv = b_h[col];
#pragma unroll
            for (int r = 0; r < 8; ++r) {
                int row = mBase + wm * 32 + i * 16 + hf * 8 + r;
                size_t idx = (size_t)row * Hsz + col;
                float vg = fast_tanh(accG[i][j][r] + bgv);   // g_new
                float vh = fast_tanh(accH[i][j][r] + bhv);   // h_cand
                float go = g_old[idx];
                float ho = h_old[idx];
                float hn = go * vh + (1.0f - go) * ho;
                h_new[idx]   = hn;
                g_new[idx]   = vg;
                hnew_bf[idx] = f2bf(hn);
            }
        }
    }
}

// ---------- stage 3: o = h_new @ lin_w.T + lin_b ----------
// lin_w is [I,H] row-major == B^T row-major -> K runs contiguous; no LDS needed.
__global__ __launch_bounds__(256)
void out_kernel(const unsigned short* __restrict__ hnbf,
                const unsigned short* __restrict__ lwbf,
                const float* __restrict__ lin_b,
                float* __restrict__ o)
{
    const int t    = threadIdx.x;
    const int lane = t & 31;
    const int wave = t >> 5;
    const int wm   = wave & 1;
    const int wn   = wave >> 1;
    const int lN   = lane & 15;
    const int hf   = lane >> 4;
    const int mBase = blockIdx.x * 64;
    const int nBase = blockIdx.y * 128;

    v8f acc[2][2] = {};

    for (int kt = 0; kt < Hsz; kt += 32) {
        Frag a[2], b[2];
#pragma unroll
        for (int i = 0; i < 2; ++i) {
            size_t base = (size_t)(mBase + wm * 32 + i * 16 + lN) * Hsz + kt;
            a[i].q[0] = *(const u32x4*)&hnbf[base + hf * 8];
            a[i].q[1] = *(const u32x4*)&hnbf[base + 16 + hf * 8];
        }
#pragma unroll
        for (int j = 0; j < 2; ++j) {
            size_t base = (size_t)(nBase + wn * 32 + j * 16 + lN) * Hsz + kt;
            b[j].q[0] = *(const u32x4*)&lwbf[base + hf * 8];
            b[j].q[1] = *(const u32x4*)&lwbf[base + 16 + hf * 8];
        }
#pragma unroll
        for (int i = 0; i < 2; ++i)
#pragma unroll
            for (int j = 0; j < 2; ++j)
                acc[i][j] = __builtin_amdgcn_wmma_f32_16x16x32_bf16(
                    false, a[i].v, false, b[j].v, (short)0, acc[i][j], false, false);
    }

#pragma unroll
    for (int i = 0; i < 2; ++i)
#pragma unroll
        for (int j = 0; j < 2; ++j) {
            int col = nBase + wn * 32 + j * 16 + lN;
            float bb = lin_b[col];
#pragma unroll
            for (int r = 0; r < 8; ++r) {
                int row = mBase + wm * 32 + i * 16 + hf * 8 + r;
                o[(size_t)row * Isz + col] = acc[i][j][r] + bb;
            }
        }
}

// ---------- host launch ----------
extern "C" void kernel_launch(void* const* d_in, const int* in_sizes, int n_in,
                              void* d_out, int out_size, void* d_ws, size_t ws_size,
                              hipStream_t stream) {
    const float* x     = (const float*)d_in[0];
    const float* h     = (const float*)d_in[1];
    const float* g     = (const float*)d_in[2];
    const float* w_xh  = (const float*)d_in[3];
    const float* w_hh  = (const float*)d_in[4];
    const float* b_h   = (const float*)d_in[5];
    const float* w_gx  = (const float*)d_in[6];
    const float* w_gh  = (const float*)d_in[7];
    const float* b_g   = (const float*)d_in[8];
    const float* lin_w = (const float*)d_in[9];
    const float* lin_b = (const float*)d_in[10];

    float* o     = (float*)d_out;
    float* h_new = o + (size_t)Bsz * Isz;
    float* g_new = h_new + (size_t)Bsz * Hsz;

    unsigned short* ws    = (unsigned short*)d_ws;
    unsigned short* xbf   = ws;
    unsigned short* hbf   = xbf   + (size_t)Bsz * Isz;
    unsigned short* wgxbf = hbf   + (size_t)Bsz * Hsz;
    unsigned short* wghbf = wgxbf + (size_t)Isz * Hsz;
    unsigned short* wxhbf = wghbf + (size_t)Hsz * Hsz;
    unsigned short* whhbf = wxhbf + (size_t)Isz * Hsz;
    unsigned short* lwbf  = whhbf + (size_t)Hsz * Hsz;
    unsigned short* hnbf  = lwbf  + (size_t)Isz * Hsz;

    // stage 1: converts
    cvt_bf16<<<1024, 256, 0, stream>>>(x,     xbf,   Bsz * Isz);
    cvt_bf16<<<1024, 256, 0, stream>>>(h,     hbf,   Bsz * Hsz);
    cvt_bf16<<<1024, 256, 0, stream>>>(w_gx,  wgxbf, Isz * Hsz);
    cvt_bf16<<<1024, 256, 0, stream>>>(w_gh,  wghbf, Hsz * Hsz);
    cvt_bf16<<<1024, 256, 0, stream>>>(w_xh,  wxhbf, Isz * Hsz);
    cvt_bf16<<<1024, 256, 0, stream>>>(w_hh,  whhbf, Hsz * Hsz);
    cvt_bf16<<<1024, 256, 0, stream>>>(lin_w, lwbf,  Isz * Hsz);

    // stage 2: fused gate cell (dual GEMM + tanh + interpolation)
    cell_kernel<<<dim3(Bsz / 64, Hsz / 128), 256, 0, stream>>>(
        xbf, hbf, wgxbf, wghbf, wxhbf, whhbf,
        b_h, b_g, g, h, h_new, g_new, hnbf);

    // stage 3: output projection
    out_kernel<<<dim3(Bsz / 64, Isz / 128), 256, 0, stream>>>(
        hnbf, lwbf, lin_b, o);
}
